// FakeModel_47227460386882
// MI455X (gfx1250) — matrix operand order repaired and checked by reading, not compiled, not run on hardware
//
#include <hip/hip_runtime.h>

// OLMoE 4-layer MoE, T=8192 tokens, D=2048, E=16 experts, top-8 (dense form).
// Compute-bound (4.4 TFLOP vs ~1GB weights): bf16 WMMA with f32 accumulate,
// f32->bf16 conversion done in-register while staging tiles to LDS.
// The 16 expert GEMMs + routing scale are fused into ONE GEMM with K'=E*D=32768:
//   out[m,n] = sum_{e,k} (route[m,e]*h[m,k]) * W_e[n,k]

#define DIM   2048
#define NEXP  16
#define TOPK  8
#define BM    128
#define BN    128
#define BK    32

typedef __attribute__((ext_vector_type(16))) __bf16 v16bf;
typedef __attribute__((ext_vector_type(8)))  float  v8f;

union FragBF { v16bf v; uint4 q[2]; };

__device__ __forceinline__ unsigned short f32_bf16(float f) {
    unsigned int u = __float_as_uint(f);
    u += 0x7FFFu + ((u >> 16) & 1u);            // round-to-nearest-even
    return (unsigned short)(u >> 16);
}

// ---------------- Router: logits -> softmax -> top-8 mask -------------------
// One block per token, 16 waves (one per expert), wave32 reduction.
__global__ __launch_bounds__(512)
void moe_router(const float* __restrict__ h, const float* __restrict__ gw,
                float* __restrict__ route) {
    const int t    = blockIdx.x;
    const int tid  = threadIdx.x;
    const int e    = tid >> 5;
    const int lane = tid & 31;
    __shared__ float logits[NEXP];

    const float* hr = h  + (size_t)t * DIM;
    const float* gr = gw + (size_t)e * DIM;
    float s = 0.f;
    for (int i = lane; i < DIM; i += 32) s += hr[i] * gr[i];
    #pragma unroll
    for (int off = 16; off >= 1; off >>= 1) s += __shfl_xor(s, off, 32);
    if (lane == 0) logits[e] = s;
    __syncthreads();

    if (tid < NEXP) {
        float l[NEXP];
        float mx = -1e30f;
        #pragma unroll
        for (int j = 0; j < NEXP; ++j) { l[j] = logits[j]; mx = fmaxf(mx, l[j]); }
        float den = 0.f;
        #pragma unroll
        for (int j = 0; j < NEXP; ++j) den += __expf(l[j] - mx);
        const int i = tid;
        int rank = 0;
        #pragma unroll
        for (int j = 0; j < NEXP; ++j)
            rank += (l[j] > l[i]) || (l[j] == l[i] && j < i);  // top_k tie: lower idx
        float w = (rank < TOPK) ? (__expf(l[i] - mx) / den) : 0.f;  // NOT renormalized
        route[(size_t)t * NEXP + i] = w;
    }
}

// ---------------- Fused expert GEMM: K' = NEXP*DIM, bf16 WMMA ---------------
__global__ __launch_bounds__(256)
void moe_expert_gemm(const float* __restrict__ h,      // [T, DIM]
                     const float* __restrict__ ew,     // [NEXP, DIM, DIM] (one layer)
                     const float* __restrict__ route,  // [T, NEXP]
                     float* __restrict__ out) {        // [T, DIM]
    __shared__ __align__(16) unsigned short As[2][BM][BK];
    __shared__ __align__(16) unsigned short Bs[2][BN][BK];

    const int tid  = threadIdx.x;
    const int lane = tid & 31;
    const int wave = tid >> 5;
    const int wm   = (wave & 1) * 64;   // 2x4 wave grid, each wave 64x32
    const int wn   = (wave >> 1) * 32;
    const int bm   = blockIdx.x * BM;
    const int bn   = blockIdx.y * BN;
    const int KT   = (NEXP * DIM) / BK; // 1024 k-tiles

    v8f acc[4][2];
    #pragma unroll
    for (int mi = 0; mi < 4; ++mi)
        #pragma unroll
        for (int ni = 0; ni < 2; ++ni)
            acc[mi][ni] = (v8f){0.f,0.f,0.f,0.f,0.f,0.f,0.f,0.f};

    // Stage one BKx(BM|BN) k-tile: read f32, apply routing weight to A rows,
    // convert to bf16 in-register, pack to LDS.
    auto stage = [&](int buf, int kt) {
        const int e  = (kt * BK) / DIM;
        const int k0 = (kt * BK) % DIM;   // BK divides DIM: tile never straddles experts
        #pragma unroll
        for (int j = 0; j < 4; ++j) {
            const int idx = tid + j * 256;
            const int row = idx >> 3;
            const int kc  = (idx & 7) * 4;
            const float4 a = *(const float4*)(h + (size_t)(bm + row) * DIM + k0 + kc);
            const float  w = route[(size_t)(bm + row) * NEXP + e];
            unsigned int lo = (unsigned int)f32_bf16(a.x * w) |
                              ((unsigned int)f32_bf16(a.y * w) << 16);
            unsigned int hi = (unsigned int)f32_bf16(a.z * w) |
                              ((unsigned int)f32_bf16(a.w * w) << 16);
            *(uint2*)&As[buf][row][kc] = make_uint2(lo, hi);

            const float4 b = *(const float4*)(ew + ((size_t)e * DIM + bn + row) * DIM + k0 + kc);
            lo = (unsigned int)f32_bf16(b.x) | ((unsigned int)f32_bf16(b.y) << 16);
            hi = (unsigned int)f32_bf16(b.z) | ((unsigned int)f32_bf16(b.w) << 16);
            *(uint2*)&Bs[buf][row][kc] = make_uint2(lo, hi);
        }
    };

    stage(0, 0);
    __syncthreads();

    for (int kt = 0; kt < KT; ++kt) {
        const int cur = kt & 1;
        if (kt + 1 < KT) stage(cur ^ 1, kt + 1);

        if (kt + 2 < KT) {  // hint next-next k-tile -> global_prefetch_b8
            const int e2 = ((kt + 2) * BK) / DIM, k2 = ((kt + 2) * BK) % DIM;
            __builtin_prefetch(h  + (size_t)(bm + (tid >> 1)) * DIM + k2 + (tid & 1) * 16, 0, 1);
            __builtin_prefetch(ew + ((size_t)e2 * DIM + bn + (tid >> 1)) * DIM + k2 + (tid & 1) * 16, 0, 1);
        }

        // Fragment loads per CDNA5 wave32 16-bit layouts.
        const int half = lane >> 4;
        FragBF a[4], b[2];
        #pragma unroll
        for (int mi = 0; mi < 4; ++mi) {
            const int row = wm + mi * 16 + (lane & 15);          // A: lane = M row
            a[mi].q[0] = *(const uint4*)&As[cur][row][half * 8];      // K 0..7 / 8..15
            a[mi].q[1] = *(const uint4*)&As[cur][row][16 + half * 8]; // K 16..23 / 24..31
        }
        #pragma unroll
        for (int ni = 0; ni < 2; ++ni) {
            const int col = wn + ni * 16 + (lane & 15);          // B: lane = N col
            b[ni].q[0] = *(const uint4*)&Bs[cur][col][half * 8];      // K 0..7 / 8..15
            b[ni].q[1] = *(const uint4*)&Bs[cur][col][16 + half * 8]; // K 16..23 / 24..31
        }

        #pragma unroll
        for (int mi = 0; mi < 4; ++mi)
            #pragma unroll
            for (int ni = 0; ni < 2; ++ni)
                acc[mi][ni] = __builtin_amdgcn_wmma_f32_16x16x32_bf16(
                    false, a[mi].v, false, b[ni].v,
                    (short)0, acc[mi][ni], false, false);

        __syncthreads();
    }

    // Epilogue: C layout — VGPR r holds M=r (lanes 0-15) / M=8+r (lanes 16-31), N=lane&15.
    #pragma unroll
    for (int mi = 0; mi < 4; ++mi)
        #pragma unroll
        for (int ni = 0; ni < 2; ++ni) {
            const int gn    = bn + wn + ni * 16 + (lane & 15);
            const int rbase = bm + wm + mi * 16 + ((lane >> 4) * 8);
            float* op = out + (size_t)rbase * DIM + gn;
            #pragma unroll
            for (int r = 0; r < 8; ++r)
                op[(size_t)r * DIM] = acc[mi][ni][r];
        }
}

// ---------------------------------------------------------------------------
extern "C" void kernel_launch(void* const* d_in, const int* in_sizes, int n_in,
                              void* d_out, int out_size, void* d_ws, size_t ws_size,
                              hipStream_t stream) {
    const float* x        = (const float*)d_in[0];  // [4, 2048, DIM]
    const float* gate_w   = (const float*)d_in[1];  // [L, NEXP, DIM]
    const float* expert_w = (const float*)d_in[2];  // [L, NEXP, DIM, DIM]
    float* out = (float*)d_out;

    const int T = in_sizes[0] / DIM;                      // 8192 tokens
    const int L = in_sizes[1] / (NEXP * DIM);             // 4 layers

    float* hbuf  = (float*)d_ws;                                    // T*DIM f32
    float* route = (float*)((char*)d_ws + (size_t)T * DIM * sizeof(float)); // T*NEXP f32

    dim3 gGemm(T / BM, DIM / BN);   // M fastest -> consecutive blocks share B n-strip in L2
    const float* src = x;
    for (int l = 0; l < L; ++l) {
        float* dst = (l & 1) ? out : hbuf;   // L even -> final layer writes d_out
        moe_router<<<T, 512, 0, stream>>>(src, gate_w + (size_t)l * NEXP * DIM, route);
        moe_expert_gemm<<<gGemm, 256, 0, stream>>>(
            src, expert_w + (size_t)l * NEXP * DIM * DIM, route, dst);
        src = dst;
    }
}
